// LSTM_Model_49426483642825
// MI455X (gfx1250) — matrix operand order repaired and checked by reading, not compiled, not run on hardware
//
#include <hip/hip_runtime.h>
#include <hip/hip_bf16.h>
#include <cstdint>

typedef __attribute__((ext_vector_type(16))) __bf16 v16bf;
typedef __attribute__((ext_vector_type(8)))  float  v8f;
typedef __attribute__((ext_vector_type(4)))  unsigned int u32x4;
typedef __attribute__((ext_vector_type(8)))  int i32x8;
typedef __attribute__((ext_vector_type(4)))  int i32x4;

#define DPAD 416   // padded input feature dim (400 -> 13 K-tiles of 32)
#define KT_IN 13
#define G4 800     // 4*H
#define NPADW 832  // Wih rows padded to 52 tiles (no guards in GEMM)

#if defined(__has_builtin)
#if __has_builtin(__builtin_amdgcn_tensor_load_to_lds) && \
    __has_builtin(__builtin_amdgcn_s_wait_tensorcnt)
#define HAVE_TDM 1
#endif
#endif

// ---------------- WMMA helpers ----------------

__device__ __forceinline__ v8f wmma_bf16(v16bf a, v16bf b, v8f c) {
  // D = A(16x32 bf16) x B(32x16 bf16) + C(16x16 f32)
  return __builtin_amdgcn_wmma_f32_16x16x32_bf16(false, a, false, b, (short)0, c,
                                                 false, false);
}

// A-matrix 16x32 bf16 layout: lane m=lane&15, khalf=lane>>4;
// elements 0..7 -> k = k0 + khalf*8 + e ; elements 8..15 -> k = k0 + 16 + khalf*8 + e
__device__ __forceinline__ v16bf load_afrag(const __bf16* rowbase, int k0, int khalf) {
  v16bf f;
  ((uint4*)&f)[0] = *(const uint4*)(rowbase + k0 + khalf * 8);
  ((uint4*)&f)[1] = *(const uint4*)(rowbase + k0 + 16 + khalf * 8);
  return f;
}

// B-matrix 32x16 bf16 layout: lane n=lane&15; elements 0..7 -> k = k0 + khalf*8 + e,
// elements 8..15 -> k = k0 + 16 + khalf*8 + e, rowbase = row n of W (W[n][k])
__device__ __forceinline__ v16bf load_bfrag(const __bf16* rowbase, int k0, int khalf) {
  v16bf f;
  ((uint4*)&f)[0] = *(const uint4*)(rowbase + k0 + khalf * 8);
  ((uint4*)&f)[1] = *(const uint4*)(rowbase + k0 + 16 + khalf * 8);
  return f;
}

__device__ __forceinline__ float fsig(float x) { return 1.0f / (1.0f + __expf(-x)); }
__device__ __forceinline__ float ftanh(float x) {
  float e = __expf(2.0f * x);
  return 1.0f - 2.0f / (e + 1.0f);
}

// ---------------- setup / conversion kernels ----------------

// Wih [8][800][400] f32 -> bf16 [8][832][416], rows permuted so column
// n' = 4*j + gate (gate order i,f,g,o); rows 800..831 zero. Combined bias (f32).
__global__ void conv_wih_kernel(const float* __restrict__ Wih,
                                const float* __restrict__ bih,
                                const float* __restrict__ bhh,
                                __bf16* __restrict__ WihB,
                                float* __restrict__ biasB) {
  int d = blockIdx.x / NPADW;
  int n = blockIdx.x % NPADW;
  __bf16* dst = WihB + ((size_t)d * NPADW + n) * DPAD;
  if (n >= 800) {
    for (int c = threadIdx.x; c < DPAD; c += blockDim.x) dst[c] = (__bf16)0.0f;
    return;
  }
  int srow = (n & 3) * 200 + (n >> 2);
  const float* src = Wih + ((size_t)d * 800 + srow) * 400;
  for (int c = threadIdx.x; c < DPAD; c += blockDim.x)
    dst[c] = (c < 400) ? (__bf16)src[c] : (__bf16)0.0f;
  if (threadIdx.x == 0)
    biasB[d * 800 + n] = bih[d * 800 + srow] + bhh[d * 800 + srow];
}

// Whh [8][800][200] f32 -> bf16 [8][800][200], same row permutation.
__global__ void conv_whh_kernel(const float* __restrict__ Whh,
                                __bf16* __restrict__ WhhB) {
  int d = blockIdx.x / 800;
  int n = blockIdx.x % 800;
  int srow = (n & 3) * 200 + (n >> 2);
  const float* src = Whh + ((size_t)d * 800 + srow) * 200;
  __bf16* dst = WhhB + ((size_t)d * 800 + n) * 200;
  for (int c = threadIdx.x; c < 200; c += blockDim.x) dst[c] = (__bf16)src[c];
}

// mlp_W1 [800][100] f32 -> transposed bf16 [2][128][416] (zero padded),
// plus b1 padded to 112 and a zero bias vector.
__global__ void conv_w1_kernel(const float* __restrict__ W1,
                               const float* __restrict__ b1,
                               __bf16* __restrict__ W1T,
                               float* __restrict__ b1p,
                               float* __restrict__ zer) {
  int h = blockIdx.x / 128;
  int n = blockIdx.x % 128;
  __bf16* dst = W1T + ((size_t)h * 128 + n) * DPAD;
  for (int k = threadIdx.x; k < DPAD; k += blockDim.x) {
    float v = (n < 100 && k < 400) ? W1[(size_t)(h * 400 + k) * 100 + n] : 0.0f;
    dst[k] = (__bf16)v;
  }
  if (blockIdx.x == 0) {
    for (int i = threadIdx.x; i < 112; i += blockDim.x) {
      b1p[i] = (i < 100) ? b1[i] : 0.0f;
      zer[i] = 0.0f;
    }
  }
}

// Embedding gather -> x0 bf16 [4096][416] (cols 0..299 word, 300..399 tag, pad 0)
__global__ void embed_kernel(const int* __restrict__ widx, const int* __restrict__ pidx,
                             const float* __restrict__ wemb, const float* __restrict__ temb,
                             __bf16* __restrict__ x0) {
  int r = blockIdx.x;
  int w = widx[r], p = pidx[r];
  __bf16* dst = x0 + (size_t)r * DPAD;
  for (int c = threadIdx.x; c < DPAD; c += blockDim.x) {
    float v;
    if (c < 300)      v = wemb[(size_t)w * 300 + c];
    else if (c < 400) v = temb[(size_t)p * 100 + (c - 300)];
    else              v = 0.0f;
    dst[c] = (__bf16)v;
  }
}

// Zero the K-pad columns (400..415) of an x buffer.
__global__ void zero_pad_kernel(__bf16* __restrict__ x) {
  int idx = blockIdx.x * blockDim.x + threadIdx.x;  // 4096*16
  int r = idx >> 4, c = idx & 15;
  if (r < 4096) x[(size_t)r * DPAD + 400 + c] = (__bf16)0.0f;
}

// ---------------- bf16 WMMA GEMM (guard-free, software pipelined) ----------------
// C[M=4096, N] = X[4096, DPAD] * W^T (+ bias[n]),  W stored as [>=ntiles*16][DPAD]
// with zero-padded rows so every wave owns 4 full tiles.
// out_mode 0: proj layout out[t][n][b]  (M rows are b*256+t, 800 valid cols)
// out_mode 1: row-major out[row][ldc]
__global__ void gemm_bf16_kernel(const __bf16* __restrict__ X,
                                 const __bf16* __restrict__ W,
                                 const float* __restrict__ bias,
                                 float* __restrict__ outp,
                                 int ntiles, int nvalid, int out_mode, int ldc) {
  const int lane = threadIdx.x & 31;
  const int wave = threadIdx.x >> 5;
  const int group = blockIdx.y * 8 + wave;
  const int nt0 = group * 4;
  if (nt0 >= ntiles) return;   // wave-uniform early exit
  const int mt = blockIdx.x;
  const int r0 = mt * 16;
  const int khalf = (lane >> 4) & 1;
  const int lm = lane & 15;
  const __bf16* arow = X + (size_t)(r0 + lm) * DPAD;
  const __bf16* brow0 = W + (size_t)((nt0 + 0) * 16 + lm) * DPAD;
  const __bf16* brow1 = W + (size_t)((nt0 + 1) * 16 + lm) * DPAD;
  const __bf16* brow2 = W + (size_t)((nt0 + 2) * 16 + lm) * DPAD;
  const __bf16* brow3 = W + (size_t)((nt0 + 3) * 16 + lm) * DPAD;

  // software pipeline: fragments for iteration kt are loaded in iteration kt-1
  v16bf a  = load_afrag(arow,  0, khalf);
  v16bf b0 = load_bfrag(brow0, 0, khalf);
  v16bf b1 = load_bfrag(brow1, 0, khalf);
  v16bf b2 = load_bfrag(brow2, 0, khalf);
  v16bf b3 = load_bfrag(brow3, 0, khalf);
  v8f acc0 = {}, acc1 = {}, acc2 = {}, acc3 = {};
#pragma unroll 1
  for (int kt = 0; kt < KT_IN; ++kt) {
    const int kn = (kt + 1 < KT_IN) ? (kt + 1) * 32 : 0;  // last iter: dummy reload
    v16bf an  = load_afrag(arow,  kn, khalf);
    v16bf b0n = load_bfrag(brow0, kn, khalf);
    v16bf b1n = load_bfrag(brow1, kn, khalf);
    v16bf b2n = load_bfrag(brow2, kn, khalf);
    v16bf b3n = load_bfrag(brow3, kn, khalf);
    acc0 = wmma_bf16(a, b0, acc0);
    acc1 = wmma_bf16(a, b1, acc1);
    acc2 = wmma_bf16(a, b2, acc2);
    acc3 = wmma_bf16(a, b3, acc3);
    a = an; b0 = b0n; b1 = b1n; b2 = b2n; b3 = b3n;
  }

  const int bidx = r0 >> 8;   // batch (mode 0)
  const int t0 = r0 & 255;    // time base (mode 0)
  const v8f* accs[4] = {&acc0, &acc1, &acc2, &acc3};
#pragma unroll
  for (int j = 0; j < 4; ++j) {
    const int n = (nt0 + j) * 16 + lm;
    if (n < nvalid) {         // divergent store mask only; no WMMA under EXEC games
      const float bv = bias[n];
      const v8f av = *accs[j];
#pragma unroll
      for (int i = 0; i < 8; ++i) {
        const float v = av[i] + bv;
        const int m = i + 8 * khalf;
        if (out_mode == 0) {
          outp[((size_t)(t0 + m) * G4 + n) * 16 + bidx] = v;
        } else {
          outp[(size_t)(r0 + m) * (size_t)ldc + n] = v;
        }
      }
    }
  }
}

// ---------------- persistent LSTM recurrence ----------------
// One 10-wave workgroup per direction; each wave owns exactly 5 of the 50
// N-tiles (no guards). Whh (800x200 bf16) staged into LDS by the Tensor Data
// Mover; h lives in LDS as the WMMA A operand; c stays in registers.
__global__ void lstm_rec_kernel(const __bf16* __restrict__ WhhL,
                                const float* __restrict__ proj0,
                                const float* __restrict__ proj1,
                                __bf16* __restrict__ xout) {
  extern __shared__ char smem[];
  __bf16* whh_s = (__bf16*)smem;               // [800][200] = 320000 B @ LDS 0
  __bf16* h_s   = (__bf16*)(smem + 320000);    // [16][200]  =   6400 B
  const int dir = blockIdx.x;
  const __bf16* whh_g = WhhL + (size_t)dir * 800 * 200;
  const float* proj = dir ? proj1 : proj0;
  const int colbase = dir * 200;
  const bool rev = (dir == 1);
  const int tid = threadIdx.x;
  const int lane = tid & 31;
  const int wave = tid >> 5;

#if HAVE_TDM
  if (wave == 0) {
    // Tensor DMA descriptor (D#): 2-D tile, 800 rows x 100 dwords, dense.
    unsigned long long ga = (unsigned long long)(uintptr_t)whh_g;
    u32x4 g0;
    g0[0] = 1u;                                  // count=1, user descriptor
    g0[1] = 0u;                                  // lds_addr = 0 (whh_s base)
    g0[2] = (unsigned)(ga & 0xFFFFFFFFu);        // global_addr[31:0]
    g0[3] = (unsigned)((ga >> 32) & 0x01FFFFFFu) // global_addr[56:32]
            | (2u << 30);                        // type=2 (image)
    i32x8 g1;
    g1[0] = (int)(2u << 16);                     // wg_mask=0, data_size=4B
    g1[1] = (int)(100u << 16);                   // tensor_dim0 = 100 (lo16)
    g1[2] = (int)(800u << 16);                   // tensor_dim1 = 800 (lo16)
    g1[3] = (int)(100u << 16);                   // tile_dim0 = 100
    g1[4] = 800;                                 // tile_dim1 = 800
    g1[5] = 100;                                 // tensor_dim0_stride = 100
    g1[6] = 0;
    g1[7] = 0;
    i32x4 g2 = {0, 0, 0, 0};
    i32x4 g3 = {0, 0, 0, 0};
    i32x8 g4 = {0, 0, 0, 0, 0, 0, 0, 0};
    __builtin_amdgcn_tensor_load_to_lds(g0, g1, g2, g3, g4, 0);
  }
#else
  {
    const uint4* src = (const uint4*)whh_g;
    uint4* dst = (uint4*)whh_s;
    for (int i = tid; i < 20000; i += 320) dst[i] = src[i];
  }
#endif
  {  // zero h
    uint4 z = make_uint4(0, 0, 0, 0);
    uint4* hz = (uint4*)h_s;
    for (int i = tid; i < 400; i += 320) hz[i] = z;
  }
#if HAVE_TDM
  if (wave == 0) __builtin_amdgcn_s_wait_tensorcnt(0);
#endif
  __syncthreads();

  const int khalf = (lane >> 4) & 1;
  const int lm = lane & 15;
  const int base4 = lane & 28;
  const __bf16* hrow = h_s + lm * 200;

  float creg[5][8];
#pragma unroll
  for (int tt = 0; tt < 5; ++tt) {
#pragma unroll
    for (int i = 0; i < 8; ++i) creg[tt][i] = 0.0f;
  }

  for (int step = 0; step < 256; ++step) {
    const int t = rev ? (255 - step) : step;

    if (step + 1 < 256) {  // warm L2->L0 for next step's gate slab
      int tn = rev ? (254 - step) : (step + 1);
      __builtin_prefetch(proj + ((size_t)tn * G4 + (wave & 7) * 100) * 16, 0, 1);
    }

    // A fragments of h (K = 200, tail tile masked)
    v16bf af[7];
#pragma unroll
    for (int kt = 0; kt < 6; ++kt) af[kt] = load_afrag(hrow, kt * 32, khalf);
    {
      v16bf f;
      uint4 z = make_uint4(0, 0, 0, 0);
      ((uint4*)&f)[0] = z; ((uint4*)&f)[1] = z;
      if (khalf == 0) ((uint4*)&f)[0] = *(const uint4*)(hrow + 192);
      af[6] = f;
    }
    __syncthreads();  // all h reads done before anyone overwrites h_s

#pragma unroll
    for (int tt = 0; tt < 5; ++tt) {
      const int tile = wave * 5 + tt;   // exactly 50 tiles over 10 waves
      const __bf16* brow = whh_s + (size_t)(tile * 16 + lm) * 200;
      v8f acc = {};
#pragma unroll
      for (int kt = 0; kt < 6; ++kt) {
        v16bf b = load_bfrag(brow, kt * 32, khalf);
        acc = wmma_bf16(af[kt], b, acc);
      }
      {
        v16bf b;
        uint4 z = make_uint4(0, 0, 0, 0);
        ((uint4*)&b)[0] = z; ((uint4*)&b)[1] = z;
        if (khalf == 0) ((uint4*)&b)[0] = *(const uint4*)(brow + 192);
        acc = wmma_bf16(af[6], b, acc);
      }
      // add precomputed input projection (layout [t][n][b])
      const int n = tile * 16 + lm;
      const float* pbase = proj + ((size_t)t * G4 + n) * 16 + khalf * 8;
      float4 p0 = *(const float4*)pbase;
      float4 p1 = *(const float4*)(pbase + 4);
      float g[8];
      g[0] = acc[0] + p0.x; g[1] = acc[1] + p0.y;
      g[2] = acc[2] + p0.z; g[3] = acc[3] + p0.w;
      g[4] = acc[4] + p1.x; g[5] = acc[5] + p1.y;
      g[6] = acc[6] + p1.z; g[7] = acc[7] + p1.w;
      // LSTM cell: gates i,f,g,o live in adjacent lanes (lane&3)
#pragma unroll
      for (int i = 0; i < 8; ++i) {
        float xi = __shfl(g[i], base4 + 0, 32);
        float xf = __shfl(g[i], base4 + 1, 32);
        float xg = __shfl(g[i], base4 + 2, 32);
        float xo = __shfl(g[i], base4 + 3, 32);
        float cn = fsig(xf) * creg[tt][i] + fsig(xi) * ftanh(xg);
        creg[tt][i] = cn;
        float hv = fsig(xo) * ftanh(cn);
        if ((lane & 3) == 0) {
          int j = tile * 4 + (lm >> 2);
          int m = i + 8 * khalf;   // batch index
          __bf16 hb = (__bf16)hv;
          h_s[m * 200 + j] = hb;
          xout[(size_t)(m * 256 + t) * DPAD + colbase + j] = hb;
        }
      }
    }
    __syncthreads();  // new h visible before next step's reads
  }
}

// ---------------- pairwise scorer ----------------
// out[s1][b][s2] = b2 + sum_r W2[r]*tanh(pa[b,s1,r] + pb[b,s2,r])  (b1 folded in pa)
__global__ void scorer_kernel(const float* __restrict__ PA, const float* __restrict__ PB,
                              const float* __restrict__ W2, const float* __restrict__ b2,
                              float* __restrict__ out) {
  extern __shared__ float sm[];
  float* pb_s = sm;               // [256][100]
  float* pa_s = sm + 25600;       // [8][100]
  float* w2_s = pa_s + 800;       // [100]
  int b = blockIdx.x >> 5;
  int grp = blockIdx.x & 31;
  int s10 = grp * 8;
  int tid = threadIdx.x;
  {
    const float* src = PB + (size_t)(b * 256 + tid) * 112;
    float* dst = pb_s + tid * 100;
    for (int r = 0; r < 100; ++r) dst[r] = src[r];
  }
  for (int idx = tid; idx < 800; idx += 256) {
    int s = idx / 100, r = idx % 100;
    pa_s[idx] = PA[(size_t)(b * 256 + s10 + s) * 112 + r];
  }
  if (tid < 100) w2_s[tid] = W2[tid];
  __syncthreads();
  float bb = b2[0];
  int s2 = tid;
  for (int s = 0; s < 8; ++s) {
    float acc = bb;
    const float* par = pa_s + s * 100;
    const float* pbr = pb_s + s2 * 100;
    for (int r = 0; r < 100; ++r) acc += w2_s[r] * ftanh(par[r] + pbr[r]);
    out[((size_t)(s10 + s) * 16 + b) * 256 + s2] = acc;
  }
}

// ---------------- host launcher ----------------

extern "C" void kernel_launch(void* const* d_in, const int* in_sizes, int n_in,
                              void* d_out, int out_size, void* d_ws, size_t ws_size,
                              hipStream_t stream) {
  const int*   widx = (const int*)d_in[0];
  const int*   pidx = (const int*)d_in[1];
  const float* wemb = (const float*)d_in[3];
  const float* temb = (const float*)d_in[4];
  const float* Wih  = (const float*)d_in[5];
  const float* Whh  = (const float*)d_in[6];
  const float* bih  = (const float*)d_in[7];
  const float* bhh  = (const float*)d_in[8];
  const float* W1   = (const float*)d_in[9];
  const float* b1   = (const float*)d_in[10];
  const float* W2   = (const float*)d_in[11];
  const float* b2   = (const float*)d_in[12];
  float* out = (float*)d_out;
  char* ws = (char*)d_ws;

  __bf16* X0    = (__bf16*)(ws + 0);          // 4096*416*2
  __bf16* X1    = (__bf16*)(ws + 3407872);
  __bf16* WIHB  = (__bf16*)(ws + 6815744);    // 8*832*416*2
  float*  BIAS  = (float*) (ws + 12353536);   // 8*800*4
  __bf16* WHHB  = (__bf16*)(ws + 12379136);   // 8*800*200*2
  float*  PROJ0 = (float*) (ws + 14939136);   // 4096*800*4, [t][n][b]
  float*  PROJ1 = (float*) (ws + 28046336);
  __bf16* W1T   = (__bf16*)(ws + 41153536);   // 2*128*416*2
  float*  B1P   = (float*) (ws + 41366528);
  float*  ZER   = (float*) (ws + 41366976);
  float*  PA    = (float*) (ws + 41367424);   // 4096*112*4
  float*  PB    = (float*) (ws + 43202432);

  conv_wih_kernel<<<dim3(8 * NPADW), dim3(128), 0, stream>>>(Wih, bih, bhh, WIHB, BIAS);
  conv_whh_kernel<<<dim3(6400), dim3(128), 0, stream>>>(Whh, WHHB);
  conv_w1_kernel<<<dim3(256), dim3(128), 0, stream>>>(W1, b1, W1T, B1P, ZER);
  embed_kernel<<<dim3(4096), dim3(128), 0, stream>>>(widx, pidx, wemb, temb, X0);
  zero_pad_kernel<<<dim3(256), dim3(256), 0, stream>>>(X1);

  const __bf16* xin = X0;
  __bf16* xo = X1;
  for (int k = 0; k < 4; ++k) {
    for (int dir = 0; dir < 2; ++dir) {
      int idx = 2 * k + dir;
      gemm_bf16_kernel<<<dim3(256, 2), dim3(256), 0, stream>>>(
          xin, WIHB + (size_t)idx * NPADW * DPAD, BIAS + idx * 800,
          dir ? PROJ1 : PROJ0, 52, 800, 0, 0);
    }
    lstm_rec_kernel<<<dim3(2), dim3(320), 326400, stream>>>(
        WHHB + (size_t)(2 * k) * 800 * 200, PROJ0, PROJ1, xo);
    const __bf16* t = xo; xo = (__bf16*)xin; xin = t;
  }
  // final features are in xin (== X0 after 4 ping-pongs)
  gemm_bf16_kernel<<<dim3(256, 1), dim3(256), 0, stream>>>(
      xin, W1T, B1P, PA, 8, 100, 1, 112);
  gemm_bf16_kernel<<<dim3(256, 1), dim3(256), 0, stream>>>(
      xin, W1T + (size_t)128 * DPAD, ZER, PB, 8, 100, 1, 112);
  scorer_kernel<<<dim3(512), dim3(256), 106000, stream>>>(PA, PB, W2, b2, out);

  (void)in_sizes; (void)n_in; (void)out_size; (void)ws_size;
}
